// SECOND_15006615733713
// MI455X (gfx1250) — matrix operand order repaired and checked by reading, not compiled, not run on hardware
//
#include <hip/hip_runtime.h>

// ---------------- constants (match reference) ----------------
#define GX 176
#define GY 200
#define GZ 10
#define BIG (GX * GY * GZ)      // 352000
#define BEV_H 200
#define BEV_W 176
#define MAX_PTS 32
#define MAX_VOX 20000
#define BN_INV 0.99950037f      // 1/sqrt(1 + 1e-3)

typedef __attribute__((ext_vector_type(16))) _Float16 v16h;
typedef __attribute__((ext_vector_type(8)))  _Float16 h8;
typedef __attribute__((ext_vector_type(8)))  float    v8f;
typedef __attribute__((ext_vector_type(4)))  unsigned u32x4;
typedef __attribute__((ext_vector_type(4)))  int      i32x4;
typedef __attribute__((ext_vector_type(8)))  int      i32x8;

// ---------------- WMMA fragment index helpers (ISA 7.12.2 layouts) ----------
// A 16x32 f16: lane<16 -> M=lane, K runs {h*8..h*8+7, 16+h*8..16+h*8+7} (h=lane>>4)
// B 32x16 f16: N=lane&15, K runs {h*8..h*8+7, 16+h*8..16+h*8+7} as for A
// C/D 16x16 f32: vgpr r -> M = r + 8*(lane>=16), N = lane&15
__device__ __forceinline__ int a_kidx(int i, int lane) {
    int j = i >> 1;
    return ((j & 4) ? 16 : 0) + (((lane >> 4) & 1) << 3) + ((j & 3) << 1) + (i & 1);
}
__device__ __forceinline__ v16h combine8(h8 lo, h8 hi) {
    return __builtin_shufflevector(lo, hi, 0, 1, 2, 3, 4, 5, 6, 7,
                                   8, 9, 10, 11, 12, 13, 14, 15);
}

// ---------------- utility fill ----------------
__global__ void fill_u32_kernel(unsigned* p, unsigned v, int n) {
    int i = blockIdx.x * blockDim.x + threadIdx.x;
    for (; i < n; i += gridDim.x * blockDim.x) p[i] = v;
}

// ---------------- weight repack (f32 -> f16, fragment-major) ----------------
// wt [CIN][COUT][2][2] -> wp [(dy*2+dx)][COUT][CIN]   (ic contiguous)
__global__ void repack_convt_w_kernel(const float* __restrict__ wt,
                                      _Float16* __restrict__ wp, int cin, int cout) {
    int i = blockIdx.x * blockDim.x + threadIdx.x;
    int total = cin * cout * 4;
    if (i >= total) return;
    int dx = i & 1, dy = (i >> 1) & 1;
    int rem = i >> 2;
    int oc = rem % cout, ic = rem / cout;
    wp[(size_t)((dy * 2 + dx) * cout + oc) * cin + ic] = (_Float16)wt[i];
}
// wc1 [64][64][3][3] -> wp [oc][tap(9)][64]  (ic contiguous)
__global__ void repack_wc1_kernel(const float* __restrict__ wc1, _Float16* __restrict__ wp) {
    int i = blockIdx.x * blockDim.x + threadIdx.x;
    if (i >= 64 * 64 * 9) return;
    int tx = i % 3, ty = (i / 3) % 3;
    int rem = i / 9;
    int ic = rem % 64, oc = rem / 64;
    wp[((size_t)oc * 9 + ty * 3 + tx) * 64 + ic] = (_Float16)wc1[i];
}
// wc2 [21][64] -> wp [32][64], rows 21..31 zero
__global__ void repack_wc2_kernel(const float* __restrict__ wc2, _Float16* __restrict__ wp) {
    int i = blockIdx.x * blockDim.x + threadIdx.x;
    if (i >= 32 * 64) return;
    int ic = i & 63, cg = i >> 6;
    wp[i] = (cg < 21) ? (_Float16)wc2[cg * 64 + ic] : (_Float16)0.0f;
}

// ---------------- voxelization ----------------
__device__ __forceinline__ int point_lin(const float* p) {
    int ix = (int)((p[0] - 0.0f)  / 0.2f);
    int iy = (int)((p[1] + 20.0f) / 0.2f);
    int iz = (int)((p[2] + 3.0f)  / 0.4f);
    if (ix < 0 || ix >= GX || iy < 0 || iy >= GY || iz < 0 || iz >= GZ) return -1;
    return ix * (GY * GZ) + iy * GZ + iz;
}

__global__ void count_kernel(const float* __restrict__ pts, int n, int* __restrict__ counts) {
    int i = blockIdx.x * blockDim.x + threadIdx.x;
    if (i >= n) return;
    int lin = point_lin(pts + (size_t)i * 7);
    if (lin >= 0) atomicAdd(&counts[lin], 1);
}

__global__ void scan_kernel(const int* __restrict__ counts, int* __restrict__ slotOf,
                            int* __restrict__ vox_lin, int* __restrict__ nvox) {
    __shared__ int sums[256];
    const int T = 256;
    int tid = threadIdx.x;
    int per = (BIG + T - 1) / T;
    int lo = tid * per;
    int hi = lo + per; if (hi > BIG) hi = BIG;
    int s = 0;
    for (int i = lo; i < hi; ++i) s += (counts[i] > 0);
    sums[tid] = s;
    __syncthreads();
    for (int off = 1; off < T; off <<= 1) {
        int v = (tid >= off) ? sums[tid - off] : 0;
        __syncthreads();
        sums[tid] += v;
        __syncthreads();
    }
    int run = sums[tid] - s;  // exclusive prefix
    for (int i = lo; i < hi; ++i) {
        if (counts[i] > 0) {
            int slot = run++;
            if (slot < MAX_VOX) { slotOf[i] = slot; vox_lin[slot] = i; }
            else slotOf[i] = -1;
        } else slotOf[i] = -1;
    }
    if (tid == T - 1) {
        int tot = sums[T - 1];
        *nvox = tot < MAX_VOX ? tot : MAX_VOX;
    }
}

__global__ void fillvox_kernel(const float* __restrict__ pts, int n,
                               const int* __restrict__ slotOf, int* __restrict__ fillc,
                               float* __restrict__ vox) {
    int i = blockIdx.x * blockDim.x + threadIdx.x;
    if (i >= n) return;
    const float* p = pts + (size_t)i * 7;
    int lin = point_lin(p);
    if (lin < 0) return;
    int s = slotOf[lin];
    if (s < 0) return;
    int pos = atomicAdd(&fillc[s], 1);
    if (pos < MAX_PTS) {
        float* dst = vox + ((size_t)s * MAX_PTS + pos) * 7;
        #pragma unroll
        for (int k = 0; k < 7; ++k) dst[k] = p[k];
    }
}

// ---------------- PFN: Linear(7,64) -> BN -> ReLU -> max over 32 rows --------
__global__ __launch_bounds__(64) void pfn_kernel(const float* __restrict__ vox,
                                                 const int* __restrict__ fillc,
                                                 const int* __restrict__ nvoxp,
                                                 const float* __restrict__ w,
                                                 const float* __restrict__ g,
                                                 const float* __restrict__ b,
                                                 float* __restrict__ feats) {
    int v = blockIdx.x;
    int tid = threadIdx.x;
    __shared__ float P[MAX_PTS][7];
    bool valid = v < *nvoxp;
    const float* src = vox + (size_t)v * MAX_PTS * 7;
    for (int i = tid; i < MAX_PTS * 7; i += 64) P[i / 7][i % 7] = valid ? src[i] : 0.0f;
    __syncthreads();
    int cnt = valid ? fillc[v] : 0;
    if (cnt > MAX_PTS) cnt = MAX_PTS;
    float inv_n = 1.0f / (float)(cnt > 1 ? cnt : 1);
    float mx = 0.f, my = 0.f, mz = 0.f;
    #pragma unroll
    for (int p = 0; p < MAX_PTS; ++p) { mx += P[p][0]; my += P[p][1]; mz += P[p][2]; }
    mx *= inv_n; my *= inv_n; mz *= inv_n;
    int c = tid;
    float wc[7];
    #pragma unroll
    for (int k = 0; k < 7; ++k) wc[k] = w[k * 64 + c];
    float gc = g[c] * BN_INV, bc = b[c];
    float best = -1e30f;
    #pragma unroll
    for (int p = 0; p < MAX_PTS; ++p) {
        float acc = (P[p][0] - mx) * wc[0] + (P[p][1] - my) * wc[1] + (P[p][2] - mz) * wc[2]
                  + P[p][3] * wc[3] + P[p][4] * wc[4] + P[p][5] * wc[5] + P[p][6] * wc[6];
        acc = fmaxf(acc * gc + bc, 0.0f);
        best = fmaxf(best, acc);
    }
    if (valid) feats[(size_t)v * 64 + c] = best;
}

// ---------------- scatter: winner = max slot id per BEV cell ----------------
__global__ void scatter_winner_kernel(const int* __restrict__ vox_lin,
                                      const int* __restrict__ nvoxp,
                                      int* __restrict__ winner) {
    int v = blockIdx.x * blockDim.x + threadIdx.x;
    if (v >= *nvoxp) return;
    int lin = vox_lin[v];
    int iy = (lin / GZ) % GY;
    int iz = lin % GZ;               // reference quirk: x index = z coord
    atomicMax(&winner[iy * BEV_W + iz], v);
}

__global__ void dense_fill_kernel(const int* __restrict__ winner,
                                  const float* __restrict__ feats,
                                  _Float16* __restrict__ dense) {
    int i = blockIdx.x * blockDim.x + threadIdx.x;
    int total = BEV_H * BEV_W * 64;
    if (i >= total) return;
    int cell = i >> 6, c = i & 63;
    int w = winner[cell];
    float v = (w >= 0) ? feats[(size_t)w * 64 + c] : 0.0f;
    dense[(size_t)c * (BEV_H * BEV_W) + cell] = (_Float16)v;
}

// ---------------- ConvTranspose2d(k=2,s=2) + bias + BN + ReLU ----------------
// Block: 128 threads (4 waves), one tile of 16 input pixels (row y, cols x0..x0+15).
// A tile staged into LDS as [ic][m] by the Tensor Data Mover (2D tile: 16 contiguous
// f16 elements per row, CIN rows at stride H*W elements). TDM is issued by wave 0,
// completion enforced with s_wait_tensorcnt + workgroup barrier.
template <int CIN, int COUT>
__global__ __launch_bounds__(128) void convt_kernel(const _Float16* __restrict__ in,
                                                    const _Float16* __restrict__ wp,
                                                    const float* __restrict__ bias,
                                                    const float* __restrict__ gbn,
                                                    const float* __restrict__ bbn,
                                                    _Float16* __restrict__ out,
                                                    int H, int W) {
    constexpr int KC = CIN / 32;
    constexpr int OCT_PER_WAVE = COUT / 64;
    int tiles_x = W >> 4;
    int y  = blockIdx.x / tiles_x;
    int x0 = (blockIdx.x % tiles_x) << 4;
    int tid = threadIdx.x, wave = tid >> 5, lane = tid & 31;

    __shared__ __align__(32) _Float16 A[CIN * 16];   // [ic][m]
    const _Float16* inrow = in + (size_t)y * W + x0;

    if (wave == 0) {
        // Tensor DMA descriptor (ISA 8.3/8.4): 2D tile, 2-byte elements
        unsigned ldsOff = (unsigned)(size_t)(void*)&A[0];   // low 32 bits = LDS byte offset
        unsigned long long ga = (unsigned long long)(size_t)inrow;
        unsigned long long s0 = (unsigned long long)H * (unsigned long long)W; // dim0 stride
        u32x4 g0;
        g0[0] = 1u;                                          // count=1, user descriptor
        g0[1] = ldsOff;                                      // lds_addr
        g0[2] = (unsigned)(ga & 0xFFFFFFFFull);              // global_addr[31:0]
        g0[3] = (unsigned)((ga >> 32) & 0x01FFFFFFull) | 0x80000000u;  // addr[56:32]|type=2
        i32x8 g1;
        unsigned tdim0 = 16u, tdim1 = (unsigned)CIN;
        g1[0] = 0x00010000;                                  // data_size=1 (2B), mask=0
        g1[1] = (int)((tdim0 & 0xFFFFu) << 16);              // tensor_dim0 lo16
        g1[2] = (int)(((tdim0 >> 16) & 0xFFFFu) | ((tdim1 & 0xFFFFu) << 16));
        g1[3] = (int)(((tdim1 >> 16) & 0xFFFFu) | (16u << 16));   // tile_dim0 = 16
        g1[4] = (int)((unsigned)CIN & 0xFFFFu);              // tile_dim1 = CIN, tile_dim2 = 0
        g1[5] = (int)(unsigned)(s0 & 0xFFFFFFFFull);         // tensor_dim0_stride lo32
        g1[6] = (int)(unsigned)((s0 >> 32) & 0xFFFFull);     // stride hi16 | dim1_stride lo16
        g1[7] = 0;
        i32x4 g2 = {0, 0, 0, 0}, g3 = {0, 0, 0, 0};
#if defined(__clang_major__) && __clang_major__ >= 23
        i32x8 g4 = {0, 0, 0, 0, 0, 0, 0, 0};
        __builtin_amdgcn_tensor_load_to_lds(g0, g1, g2, g3, g4, 0);
#else
        __builtin_amdgcn_tensor_load_to_lds(g0, g1, g2, g3, 0);
#endif
        __builtin_amdgcn_s_wait_tensorcnt(0);
    }
    __builtin_prefetch(inrow + 16, 0, 0);     // global_prefetch_b8: next tile's row
    __syncthreads();

    int mrow = lane & 15;
    v16h a[KC];
    #pragma unroll
    for (int kc = 0; kc < KC; ++kc)
        #pragma unroll
        for (int i = 0; i < 16; ++i)
            a[kc][i] = A[(kc * 32 + a_kidx(i, lane)) * 16 + mrow];

    int H2 = H * 2, W2 = W * 2;
    #pragma unroll
    for (int ot = 0; ot < OCT_PER_WAVE; ++ot) {
        int oc = (wave * OCT_PER_WAVE + ot) * 16 + (lane & 15);
        float bo = bias[oc];
        float gg = gbn[oc] * BN_INV, bb = bbn[oc];
        #pragma unroll
        for (int ph = 0; ph < 4; ++ph) {
            int dy = ph >> 1, dx = ph & 1;
            v8f acc = {};
            const _Float16* wrow = wp + (size_t)(ph * COUT + oc) * CIN
                                      + (((lane >> 4) & 1) << 3);
            #pragma unroll
            for (int kc = 0; kc < KC; ++kc) {
                // B fragment: k runs {8h..8h+7, 16+8h..16+8h+7} -> 2x global b128
                h8 blo = *(const h8*)(wrow + kc * 32);
                h8 bhi = *(const h8*)(wrow + kc * 32 + 16);
                acc = __builtin_amdgcn_wmma_f32_16x16x32_f16(false, a[kc], false,
                                                             combine8(blo, bhi),
                                                             (short)0, acc, false, false);
            }
            #pragma unroll
            for (int r = 0; r < 8; ++r) {
                int m = r + (((lane >> 4) & 1) << 3);
                float v = fmaxf((acc[r] + bo) * gg + bb, 0.0f);
                out[(size_t)oc * H2 * W2 + (size_t)(2 * y + dy) * W2 + (2 * (x0 + m) + dx)] =
                    (_Float16)v;
            }
        }
    }
}

// ---------------- fused head: conv3x3+BN+ReLU -> conv1x1, writes [H*W, 21] ----
__global__ __launch_bounds__(128) void head_kernel(const _Float16* __restrict__ in,
                                                   const _Float16* __restrict__ wp1,
                                                   const float* __restrict__ bc1,
                                                   const float* __restrict__ g3,
                                                   const float* __restrict__ b3,
                                                   const _Float16* __restrict__ wp2,
                                                   const float* __restrict__ bc2,
                                                   float* __restrict__ out) {
    const int H = 800, W = 704;
    int tiles_x = W >> 4;                        // 44
    int y  = blockIdx.x / tiles_x;
    int x0 = (blockIdx.x % tiles_x) << 4;
    int tid = threadIdx.x, wave = tid >> 5, lane = tid & 31;

    __shared__ __align__(16) _Float16 IN[3][18][64];   // halo tile, ic contiguous
    __shared__ __align__(16) _Float16 H3[16][64];      // conv3x3 result (f16)

    for (int i = tid; i < 3 * 18 * 64; i += 128) {
        int ic = i & 63;
        int rem = i >> 6;
        int c = rem % 18, ry = rem / 18;
        int gy = y + ry - 1, gx = x0 + c - 1;
        _Float16 v = (_Float16)0.0f;
        if (gy >= 0 && gy < H && gx >= 0 && gx < W)
            v = in[(size_t)ic * H * W + (size_t)gy * W + gx];
        IN[ry][c][ic] = v;
    }
    __syncthreads();

    int mrow = lane & 15;
    int h8sel = (lane >> 4) & 1;
    int oc = wave * 16 + (lane & 15);            // each wave owns one oc tile of 16
    v8f acc = {};
    // K = 9 taps x 64 ic = 576 -> 18 k-steps of 32 (tap-major)
    for (int ks = 0; ks < 18; ++ks) {
        int tap = ks >> 1, ty = tap / 3, tx = tap % 3;
        int icbase = (ks & 1) * 32;
        const _Float16* arow = &IN[ty][mrow + tx][icbase + h8sel * 8];
        h8 alo = *(const h8*)(arow);             // K run 0..7
        h8 ahi = *(const h8*)(arow + 16);        // K run 16..23
        const _Float16* brow = wp1 + ((size_t)oc * 9 + tap) * 64 + icbase + h8sel * 8;
        h8 blo = *(const h8*)(brow);
        h8 bhi = *(const h8*)(brow + 16);
        acc = __builtin_amdgcn_wmma_f32_16x16x32_f16(false, combine8(alo, ahi), false,
                                                     combine8(blo, bhi),
                                                     (short)0, acc, false, false);
    }
    {   // conv3x3 epilogue: bias + BN + ReLU -> LDS (f16)
        float bo = bc1[oc], gg = g3[oc] * BN_INV, bb = b3[oc];
        #pragma unroll
        for (int r = 0; r < 8; ++r) {
            int m = r + (h8sel << 3);
            H3[m][oc] = (_Float16)fmaxf((acc[r] + bo) * gg + bb, 0.0f);
        }
    }
    __syncthreads();

    // conv1x1: 21 output channels padded to 32 -> waves 0,1 (wave-uniform branch)
    if (wave < 2) {
        int cg = wave * 16 + (lane & 15);
        v8f a2 = {};
        #pragma unroll
        for (int kc = 0; kc < 2; ++kc) {
            const _Float16* arow = &H3[mrow][kc * 32 + h8sel * 8];
            h8 alo = *(const h8*)(arow);
            h8 ahi = *(const h8*)(arow + 16);
            const _Float16* brow = wp2 + (size_t)cg * 64 + kc * 32 + h8sel * 8;
            h8 blo = *(const h8*)(brow);
            h8 bhi = *(const h8*)(brow + 16);
            a2 = __builtin_amdgcn_wmma_f32_16x16x32_f16(false, combine8(alo, ahi), false,
                                                        combine8(blo, bhi),
                                                        (short)0, a2, false, false);
        }
        if (cg < 21) {
            float bo = bc2[cg];
            #pragma unroll
            for (int r = 0; r < 8; ++r) {
                int m = r + (h8sel << 3);
                size_t p = (size_t)y * W + (size_t)(x0 + m);
                out[p * 21 + cg] = a2[r] + bo;
            }
        }
    }
}

// ---------------- host-side orchestration ----------------
extern "C" void kernel_launch(void* const* d_in, const int* in_sizes, int n_in,
                              void* d_out, int out_size, void* d_ws, size_t ws_size,
                              hipStream_t stream) {
    (void)n_in; (void)out_size; (void)ws_size;
    const float* points = (const float*)d_in[0];
    const float* w_pfn  = (const float*)d_in[1];
    const float* g_pfn  = (const float*)d_in[2];
    const float* b_pfn  = (const float*)d_in[3];
    const float* wt1    = (const float*)d_in[4];
    const float* bt1    = (const float*)d_in[5];
    const float* g1     = (const float*)d_in[6];
    const float* b1     = (const float*)d_in[7];
    const float* wt2    = (const float*)d_in[8];
    const float* bt2    = (const float*)d_in[9];
    const float* g2     = (const float*)d_in[10];
    const float* b2     = (const float*)d_in[11];
    const float* wc1    = (const float*)d_in[12];
    const float* bc1    = (const float*)d_in[13];
    const float* g3     = (const float*)d_in[14];
    const float* b3     = (const float*)d_in[15];
    const float* wc2    = (const float*)d_in[16];
    const float* bc2    = (const float*)d_in[17];
    int N = in_sizes[0] / 7;

    char* ws = (char*)d_ws;
    size_t off = 0;
    auto alloc = [&](size_t bytes) -> void* {
        off = (off + 255) & ~(size_t)255;
        void* p = ws + off;
        off += bytes;
        return p;
    };
    int*       counts  = (int*)      alloc((size_t)BIG * 4);
    int*       slotOf  = (int*)      alloc((size_t)BIG * 4);
    int*       vox_lin = (int*)      alloc((size_t)MAX_VOX * 4);
    int*       fillc   = (int*)      alloc((size_t)MAX_VOX * 4);
    int*       nvox    = (int*)      alloc(64);
    int*       winner  = (int*)      alloc((size_t)BEV_H * BEV_W * 4);
    float*     vox     = (float*)    alloc((size_t)MAX_VOX * MAX_PTS * 7 * 4);
    float*     feats   = (float*)    alloc((size_t)MAX_VOX * 64 * 4);
    _Float16*  dense   = (_Float16*) alloc((size_t)64 * BEV_H * BEV_W * 2);
    _Float16*  h1      = (_Float16*) alloc((size_t)128 * 400 * 352 * 2);
    _Float16*  h2      = (_Float16*) alloc((size_t)64 * 800 * 704 * 2);
    _Float16*  wp1     = (_Float16*) alloc((size_t)4 * 128 * 64 * 2);    // convT1
    _Float16*  wp2     = (_Float16*) alloc((size_t)4 * 64 * 128 * 2);    // convT2
    _Float16*  wpc1    = (_Float16*) alloc((size_t)64 * 9 * 64 * 2);     // conv3x3
    _Float16*  wpc2    = (_Float16*) alloc((size_t)32 * 64 * 2);         // conv1x1 padded

    // init scratch
    fill_u32_kernel<<<512, 256, 0, stream>>>((unsigned*)counts, 0u, BIG);
    fill_u32_kernel<<<64, 256, 0, stream>>>((unsigned*)fillc, 0u, MAX_VOX);
    fill_u32_kernel<<<64, 256, 0, stream>>>((unsigned*)winner, 0xFFFFFFFFu, BEV_H * BEV_W);
    fill_u32_kernel<<<2048, 256, 0, stream>>>((unsigned*)vox, 0u, MAX_VOX * MAX_PTS * 7);

    // weight repack (f32 -> fragment-major f16)
    repack_convt_w_kernel<<<(64 * 128 * 4 + 255) / 256, 256, 0, stream>>>(wt1, wp1, 64, 128);
    repack_convt_w_kernel<<<(128 * 64 * 4 + 255) / 256, 256, 0, stream>>>(wt2, wp2, 128, 64);
    repack_wc1_kernel<<<(64 * 64 * 9 + 255) / 256, 256, 0, stream>>>(wc1, wpc1);
    repack_wc2_kernel<<<(32 * 64 + 255) / 256, 256, 0, stream>>>(wc2, wpc2);

    // voxelize
    count_kernel<<<(N + 255) / 256, 256, 0, stream>>>(points, N, counts);
    scan_kernel<<<1, 256, 0, stream>>>(counts, slotOf, vox_lin, nvox);
    fillvox_kernel<<<(N + 255) / 256, 256, 0, stream>>>(points, N, slotOf, fillc, vox);

    // PFN + scatter to dense BEV
    pfn_kernel<<<MAX_VOX, 64, 0, stream>>>(vox, fillc, nvox, w_pfn, g_pfn, b_pfn, feats);
    scatter_winner_kernel<<<(MAX_VOX + 255) / 256, 256, 0, stream>>>(vox_lin, nvox, winner);
    dense_fill_kernel<<<(BEV_H * BEV_W * 64 + 255) / 256, 256, 0, stream>>>(winner, feats, dense);

    // ConvT stack (WMMA f16 phase-GEMMs, TDM-staged A tiles)
    convt_kernel<64, 128><<<200 * (176 / 16), 128, 0, stream>>>(dense, wp1, bt1, g1, b1, h1, 200, 176);
    convt_kernel<128, 64><<<400 * (352 / 16), 128, 0, stream>>>(h1, wp2, bt2, g2, b2, h2, 400, 352);

    // Fused head (conv3x3+BN+ReLU + conv1x1), WMMA implicit GEMM
    head_kernel<<<800 * (704 / 16), 128, 0, stream>>>(h2, wpc1, bc1, g3, b3, wpc2, bc2, (float*)d_out);
}
